// MelToAudio_81243601371643
// MI455X (gfx1250) — matrix operand order repaired and checked by reading, not compiled, not run on hardware
//
#include <hip/hip_runtime.h>
#include <math.h>

typedef __attribute__((ext_vector_type(16))) _Float16 v16h;
typedef __attribute__((ext_vector_type(8)))  _Float16 v8h;
typedef __attribute__((ext_vector_type(8)))  float    v8f;
typedef int v4i __attribute__((vector_size(16)));   // pointee type the async builtin expects

#define NFFT   1024
#define HOP    256
#define NB     513
#define NBP    528                 // bins padded to mult of 16
#define KINV   1056                // 2*NBP : K of inverse-DFT GEMM (mult of 32)
#define MFWD   1088                // fwd-DFT GEMM M padded to mult of 64 (>= 1041)
#define NMELS  80
#define BATCH  8
#define TFRM   1024
#define NCOL   (BATCH*TFRM)        // 8192 GEMM columns
#define PADLEN (NFFT + HOP*(TFRM-1))   // 262912
#define AUDLEN (HOP*(TFRM-1))          // 261888
#define NITER  32
#define GLBETA (0.99f/1.99f)
#define SPECN  (BATCH*NB*TFRM)     // 4202496
#define TWO_PI 6.283185307179586

// ---------------- CDNA5 async global->LDS helpers ----------------
// GLOBAL_LOAD_ASYNC_TO_LDS_B128 (ASYNCcnt-tracked, memory engine writes LDS
// directly); falls back to a synchronous 16B copy if the builtin is absent.
__device__ __forceinline__ void async_copy16(const _Float16* g, _Float16* l) {
#if defined(__gfx1250__) && __has_builtin(__builtin_amdgcn_global_load_async_to_lds_b128)
  __builtin_amdgcn_global_load_async_to_lds_b128(
      (__attribute__((address_space(1))) v4i*)g,
      (__attribute__((address_space(3))) v4i*)l, 0, 0);
#else
  *(int4*)l = *(const int4*)g;
#endif
}
__device__ __forceinline__ void wait_async3() {
#if defined(__gfx1250__)
  asm volatile("s_wait_asynccnt 0x3" ::: "memory");
#endif
}
__device__ __forceinline__ void wait_async0() {
#if defined(__gfx1250__)
  asm volatile("s_wait_asynccnt 0x0" ::: "memory");
#endif
}

// ---------------- init kernels ----------------

__global__ void win_kernel(float* __restrict__ w) {
  int n = blockIdx.x*blockDim.x + threadIdx.x;
  if (n < NFFT) w[n] = (float)(0.5 - 0.5*cos(TWO_PI*(double)n/(double)NFFT));
}

// wsq[p] = sum over frames f of window[p - f*HOP]^2  (hann overlap-add normalizer)
__global__ void wsq_kernel(const float* __restrict__ w, float* __restrict__ wsq) {
  int p = blockIdx.x*blockDim.x + threadIdx.x;
  if (p >= PADLEN) return;
  int fmin = (p - (NFFT-1) + (HOP-1)) >> 8;  if (fmin < 0) fmin = 0;
  int fmax = p >> 8;                          if (fmax > TFRM-1) fmax = TFRM-1;
  float s = 0.f;
  for (int f = fmin; f <= fmax; ++f) { float v = w[p - f*HOP]; s += v*v; }
  wsq[p] = s;
}

// Forward DFT table, window folded in.  Row-major [MFWD][NFFT].
__global__ void fwd_table_kernel(const float* __restrict__ w, _Float16* __restrict__ C) {
  int idx = blockIdx.x*blockDim.x + threadIdx.x;
  if (idx >= MFWD*NFFT) return;
  int r = idx / NFFT, n = idx % NFFT;
  float v = 0.f;
  if (r < NB) {
    int ph = (r*n) & (NFFT-1);
    v = w[n] * (float)cos(TWO_PI*(double)ph/(double)NFFT);
  } else if (r >= NBP && r < NBP+NB) {
    int k = r - NBP;
    int ph = (k*n) & (NFFT-1);
    v = -w[n] * (float)sin(TWO_PI*(double)ph/(double)NFFT);
  }
  C[idx] = (_Float16)v;
}

// Inverse DFT table (irfft expansion), window + 1/N + hermitian factor folded in.
__global__ void inv_table_kernel(const float* __restrict__ w, _Float16* __restrict__ C) {
  int idx = blockIdx.x*blockDim.x + threadIdx.x;
  if (idx >= NFFT*KINV) return;
  int n = idx / KINV, c = idx % KINV;
  float v = 0.f;
  if (c < NB) {
    int k = c;
    float sk = ((k == 0) || (k == NFFT/2)) ? (1.0f/NFFT) : (2.0f/NFFT);
    int ph = (k*n) & (NFFT-1);
    v = sk * w[n] * (float)cos(TWO_PI*(double)ph/(double)NFFT);
  } else if (c >= NBP && c < NBP+NB) {
    int k = c - NBP;
    if (k != 0 && k != NFFT/2) {
      float sk = 2.0f/NFFT;
      int ph = (k*n) & (NFFT-1);
      v = -sk * w[n] * (float)sin(TWO_PI*(double)ph/(double)NFFT);
    }
  }
  C[idx] = (_Float16)v;
}

__global__ void exp_kernel(const float* __restrict__ x, float* __restrict__ e) {
  int i = blockIdx.x*blockDim.x + threadIdx.x;
  if (i < BATCH*NMELS*TFRM) e[i] = expf(x[i]);
}

// mag[b,k,t] = | sum_m mel_inv[k,m] * E[b,m,t] |
__global__ void mag_kernel(const float* __restrict__ minv, const float* __restrict__ E,
                           float* __restrict__ mag) {
  int i = blockIdx.x*blockDim.x + threadIdx.x;
  if (i >= SPECN) return;
  int b = i / (NB*TFRM);
  int rem = i % (NB*TFRM);
  int k = rem / TFRM, t = rem % TFRM;
  const float* Eb = E + (size_t)b*NMELS*TFRM + t;
  const float* Mk = minv + (size_t)k*NMELS;
  float acc = 0.f;
  #pragma unroll 4
  for (int m = 0; m < NMELS; ++m) acc = fmaf(Mk[m], Eb[(size_t)m*TFRM], acc);
  mag[i] = fabsf(acc);
}

__global__ void angle_kernel(const float* __restrict__ u,
                             float* __restrict__ aRe, float* __restrict__ aIm) {
  int i = blockIdx.x*blockDim.x + threadIdx.x;
  if (i >= SPECN) return;
  float ang = 6.2831853f * u[i];
  aRe[i] = cosf(ang);
  aIm[i] = sinf(ang);
}

// ---------------- per-iteration glue kernels ----------------

// Pack S (GEMM B, column-major storage [col][KINV]) = mag * e^{i theta}, f16.
__global__ void pack_kernel(const float* __restrict__ mag,
                            const float* __restrict__ aRe, const float* __restrict__ aIm,
                            _Float16* __restrict__ S) {
  int i = blockIdx.x*blockDim.x + threadIdx.x;
  if (i >= NCOL*NBP) return;
  int k = i % NBP, col = i / NBP;
  int b = col >> 10, t = col & (TFRM-1);
  _Float16 re = (_Float16)0.f, im = (_Float16)0.f;
  if (k < NB) {
    size_t mi = (size_t)b*NB*TFRM + (size_t)k*TFRM + t;
    float m = mag[mi];
    re = (_Float16)(m * aRe[mi]);
    im = (_Float16)(m * aIm[mi]);
  }
  S[(size_t)col*KINV + k]       = re;
  S[(size_t)col*KINV + NBP + k] = im;
}

// audioC[b,i] = audioP[b, i+512] / max(wsq[i+512], 1e-8)
__global__ void norm_kernel(const float* __restrict__ audioP, const float* __restrict__ wsq,
                            float* __restrict__ outC) {
  int i = blockIdx.x*blockDim.x + threadIdx.x;
  if (i >= BATCH*AUDLEN) return;
  int b = i / AUDLEN, j = i % AUDLEN;
  outC[i] = audioP[(size_t)b*PADLEN + j + NFFT/2] / fmaxf(wsq[j + NFFT/2], 1e-8f);
}

// Fr (GEMM B for fwd DFT, [col][NFFT] f16) = reflect-padded framed audio.
__global__ void frame_kernel(const float* __restrict__ audioC, _Float16* __restrict__ Fr) {
  int i = blockIdx.x*blockDim.x + threadIdx.x;
  if (i >= NCOL*NFFT) return;
  int n = i & (NFFT-1), col = i >> 10;
  int b = col >> 10, f = col & (TFRM-1);
  int q = f*HOP + n - NFFT/2;
  if (q < 0) q = -q;
  if (q >= AUDLEN) q = 2*AUDLEN - 2 - q;
  Fr[i] = (_Float16)audioC[(size_t)b*AUDLEN + q];
}

// new_ang = (reb - beta*reb_prev) normalized to unit magnitude
__global__ void phase_kernel(const float* __restrict__ nR, const float* __restrict__ nI,
                             const float* __restrict__ pR, const float* __restrict__ pI,
                             float* __restrict__ aRe, float* __restrict__ aIm) {
  int i = blockIdx.x*blockDim.x + threadIdx.x;
  if (i >= SPECN) return;
  int b = i / (NB*TFRM);
  int rem = i % (NB*TFRM);
  int k = rem / TFRM, t = rem % TFRM;
  size_t ci = (size_t)k*NCOL + (size_t)b*TFRM + t;   // reb layout [k][col]
  float xr = nR[ci] - GLBETA*pR[ci];
  float xi = nI[ci] - GLBETA*pI[ci];
  float r = sqrtf(xr*xr + xi*xi) + 1e-16f;
  aRe[i] = xr / r;
  aIm[i] = xi / r;
}

// ---------------- WMMA GEMM ----------------
// D[M x 8192] = A[M x KDIM] * B, with B stored column-major ([col][KDIM]).
// Block tile 64(M) x 128(N), 8 waves (4x2), each wave: 16x64 via 4 accumulators.
// Double-buffered LDS; staging via async global->LDS copies (ASYNCcnt pipeline).
// EPI 0: iDFT -> fused overlap-add (atomic f32) into padded audio.
// EPI 1: fwd DFT -> scatter rows into rebRe / rebIm.
template<int KDIM, int EPI>
__global__ __launch_bounds__(256)
void gemm_wmma(const _Float16* __restrict__ A, const _Float16* __restrict__ Bcm,
               float* __restrict__ out0, float* __restrict__ out1) {
  constexpr int BK = 32;
  constexpr int NK = KDIM / BK;
  __shared__ __align__(16) _Float16 As[2][64*BK];
  __shared__ __align__(16) _Float16 Bs[2][128*BK];
  const int tid  = threadIdx.x;
  const int lane = tid & 31;
  const int wid  = tid >> 5;
  const int waveM = wid & 3;
  const int waveN = wid >> 2;
  const int l15  = lane & 15;
  const int hi   = lane >> 4;
  const int m0 = blockIdx.y * 64;
  const int n0 = blockIdx.x * 128;

  const int arow = tid >> 2, acol = (tid & 3) * 8;   // 8 halves / thread
  const int brow = tid >> 1, bcol = (tid & 1) * 16;  // 16 halves / thread
  const _Float16* gA = &A  [(size_t)(m0 + arow)*KDIM + acol];
  const _Float16* gB = &Bcm[(size_t)(n0 + brow)*KDIM + bcol];

  // stage tile k into LDS buffer `buf` (3 x 16B per thread)
  auto issue = [&](int k, int buf) {
    async_copy16(gA + k*BK,     &As[buf][arow*BK + acol]);
    async_copy16(gB + k*BK,     &Bs[buf][brow*BK + bcol]);
    async_copy16(gB + k*BK + 8, &Bs[buf][brow*BK + bcol + 8]);
  };

  v8f acc[4] = {v8f{}, v8f{}, v8f{}, v8f{}};

  issue(0, 0);
  for (int k = 0; k < NK; ++k) {
    const int cur = k & 1;
    if (k + 1 < NK) { issue(k + 1, cur ^ 1); wait_async3(); }  // tile k landed, k+1 in flight
    else            { wait_async0(); }
    __syncthreads();   // all waves' tile-k data visible in LDS

    // A fragment: lane l15 = row M, halves i: K = (i&7) + 16*(i>>3) + 8*hi
    union { v16h v; v8h h[2]; } af;
    const int ab = (waveM*16 + l15)*BK + 8*hi;
    af.h[0] = *(const v8h*)(&As[cur][ab]);
    af.h[1] = *(const v8h*)(&As[cur][ab + 16]);

    #pragma unroll
    for (int ni = 0; ni < 4; ++ni) {
      // B fragment: lane l15 = col N, halves i: K = (i&7) + 16*(i>>3) + 8*hi  (Bs is [n][k])
      union { v16h v; v8h h[2]; } bf;
      const int bb = (waveN*64 + ni*16 + l15)*BK + 8*hi;
      bf.h[0] = *(const v8h*)(&Bs[cur][bb]);
      bf.h[1] = *(const v8h*)(&Bs[cur][bb + 16]);
      acc[ni] = __builtin_amdgcn_wmma_f32_16x16x32_f16(
          false, af.v, false, bf.v, (short)0, acc[ni], false, false);
    }
    __syncthreads();   // everyone done reading buf `cur` before it is re-filled
  }

  #pragma unroll
  for (int ni = 0; ni < 4; ++ni) {
    #pragma unroll
    for (int r = 0; r < 8; ++r) {
      float d = acc[ni][r];
      int m   = m0 + waveM*16 + hi*8 + r;       // D row = r + 8*(lane>=16)
      int col = n0 + waveN*64 + ni*16 + l15;    // D col = lane&15
      if (EPI == 0) {
        int b = col >> 10, f = col & (TFRM-1);
        atomicAdd(&out0[(size_t)b*PADLEN + f*HOP + m], d);   // overlap-add
      } else {
        if (m < NB)                       out0[(size_t)m*NCOL + col] = d;
        else if (m >= NBP && m < NBP+NB)  out1[(size_t)(m-NBP)*NCOL + col] = d;
      }
    }
  }
}

// ---------------- host ----------------

extern "C" void kernel_launch(void* const* d_in, const int* in_sizes, int n_in,
                              void* d_out, int out_size, void* d_ws, size_t ws_size,
                              hipStream_t stream) {
  (void)in_sizes; (void)n_in; (void)out_size; (void)ws_size;
  const float* mel_spec = (const float*)d_in[0];
  const float* mel_inv  = (const float*)d_in[1];
  const float* angles0  = (const float*)d_in[2];
  float* out = (float*)d_out;

  char* ws = (char*)d_ws;
  size_t off = 0;
  auto alloc = [&](size_t bytes) -> void* {
    void* p = ws + off;
    off = (off + bytes + 255) & ~(size_t)255;
    return p;
  };

  float*    win    = (float*)   alloc(NFFT*sizeof(float));
  float*    wsq    = (float*)   alloc(PADLEN*sizeof(float));
  _Float16* Cfwd   = (_Float16*)alloc((size_t)MFWD*NFFT*sizeof(_Float16));
  _Float16* Cinv   = (_Float16*)alloc((size_t)NFFT*KINV*sizeof(_Float16));
  float*    E      = (float*)   alloc((size_t)BATCH*NMELS*TFRM*sizeof(float));
  float*    mag    = (float*)   alloc((size_t)SPECN*sizeof(float));
  float*    aRe    = (float*)   alloc((size_t)SPECN*sizeof(float));
  float*    aIm    = (float*)   alloc((size_t)SPECN*sizeof(float));
  float*    rebRe[2], *rebIm[2];
  rebRe[0] = (float*)alloc((size_t)NB*NCOL*sizeof(float));
  rebRe[1] = (float*)alloc((size_t)NB*NCOL*sizeof(float));
  rebIm[0] = (float*)alloc((size_t)NB*NCOL*sizeof(float));
  rebIm[1] = (float*)alloc((size_t)NB*NCOL*sizeof(float));
  _Float16* S      = (_Float16*)alloc((size_t)NCOL*KINV*sizeof(_Float16));
  _Float16* Fr     = (_Float16*)alloc((size_t)NCOL*NFFT*sizeof(_Float16));
  float*    audioP = (float*)   alloc((size_t)BATCH*PADLEN*sizeof(float));
  float*    audioC = (float*)   alloc((size_t)BATCH*AUDLEN*sizeof(float));

  const int TB = 256;
  // one-time setup (per launch; deterministic)
  win_kernel      <<<(NFFT+TB-1)/TB, TB, 0, stream>>>(win);
  wsq_kernel      <<<(PADLEN+TB-1)/TB, TB, 0, stream>>>(win, wsq);
  fwd_table_kernel<<<((size_t)MFWD*NFFT+TB-1)/TB, TB, 0, stream>>>(win, Cfwd);
  inv_table_kernel<<<((size_t)NFFT*KINV+TB-1)/TB, TB, 0, stream>>>(win, Cinv);
  exp_kernel      <<<(BATCH*NMELS*TFRM+TB-1)/TB, TB, 0, stream>>>(mel_spec, E);
  mag_kernel      <<<(SPECN+TB-1)/TB, TB, 0, stream>>>(mel_inv, E, mag);
  angle_kernel    <<<(SPECN+TB-1)/TB, TB, 0, stream>>>(angles0, aRe, aIm);
  (void)hipMemsetAsync(rebRe[1], 0, (size_t)NB*NCOL*sizeof(float), stream);
  (void)hipMemsetAsync(rebIm[1], 0, (size_t)NB*NCOL*sizeof(float), stream);

  dim3 gInv(NCOL/128, NFFT/64);   // iDFT GEMM grid
  dim3 gFwd(NCOL/128, MFWD/64);   // fwd DFT GEMM grid

  for (int it = 0; it < NITER; ++it) {
    int ni = it & 1, pi = 1 - ni;   // ping-pong reb buffers
    pack_kernel<<<((size_t)NCOL*NBP+TB-1)/TB, TB, 0, stream>>>(mag, aRe, aIm, S);
    (void)hipMemsetAsync(audioP, 0, (size_t)BATCH*PADLEN*sizeof(float), stream);
    gemm_wmma<KINV, 0><<<gInv, TB, 0, stream>>>(Cinv, S, audioP, nullptr);
    norm_kernel <<<((size_t)BATCH*AUDLEN+TB-1)/TB, TB, 0, stream>>>(audioP, wsq, audioC);
    frame_kernel<<<((size_t)NCOL*NFFT+TB-1)/TB, TB, 0, stream>>>(audioC, Fr);
    gemm_wmma<NFFT, 1><<<gFwd, TB, 0, stream>>>(Cfwd, Fr, rebRe[ni], rebIm[ni]);
    phase_kernel<<<(SPECN+TB-1)/TB, TB, 0, stream>>>(rebRe[ni], rebIm[ni],
                                                     rebRe[pi], rebIm[pi], aRe, aIm);
  }

  // final iSTFT with converged angles -> d_out
  pack_kernel<<<((size_t)NCOL*NBP+TB-1)/TB, TB, 0, stream>>>(mag, aRe, aIm, S);
  (void)hipMemsetAsync(audioP, 0, (size_t)BATCH*PADLEN*sizeof(float), stream);
  gemm_wmma<KINV, 0><<<gInv, TB, 0, stream>>>(Cinv, S, audioP, nullptr);
  norm_kernel<<<((size_t)BATCH*AUDLEN+TB-1)/TB, TB, 0, stream>>>(audioP, wsq, out);
}
